// PerceiverAttention_14233521619502
// MI455X (gfx1250) — matrix operand order repaired and checked by reading, not compiled, not run on hardware
//
#include <hip/hip_runtime.h>

// ---------------------------------------------------------------------------
// PerceiverAttention forward for MI455X (gfx1250, wave32, WMMA).
// All matmuls on v_wmma_f32_16x16x32_bf16; LDS tiles staged with CDNA5 async
// copies (global_load_async_to_lds_b128, ASYNCcnt); all fragment feeds are
// 2 x ds_load_b128; weights and V pre-transposed so no scalar gathers.
// ---------------------------------------------------------------------------

typedef __attribute__((ext_vector_type(16))) __bf16        v16bf;
typedef __attribute__((ext_vector_type(8)))  float         v8f;
typedef __attribute__((ext_vector_type(4)))  unsigned int  u32x4;
typedef __attribute__((ext_vector_type(4)))  float         f32x4;

#define D_MODEL 1024
#define HEADS   16
#define DH      64
#define B_LAT   9
#define B_X     8
#define N_KV    256
#define N_Q     64
#define ROWS_Q  (B_LAT * N_Q)            // 576
#define ROWS_KV (B_X * (N_KV + N_Q))     // 2560

__device__ __forceinline__ unsigned short f2bf(float f) {
    return __builtin_bit_cast(unsigned short, static_cast<__bf16>(f));
}

// LDS byte offset of a __shared__ object: generic pointers to LDS carry the
// aperture in the high 32 bits only (ISA 10.2 aperture check), so the low 32
// bits are the wave-relative LDS address the async-copy VDST operand wants.
__device__ __forceinline__ unsigned lds_off(const void* p) {
    return (unsigned)(size_t)p;
}

// CDNA5 async copy: 16 bytes global -> LDS per lane, tracked on ASYNCcnt.
__device__ __forceinline__ void async_copy_b128(unsigned lds, const void* gptr) {
    asm volatile("global_load_async_to_lds_b128 %0, %1, off"
                 :: "v"(lds), "v"((unsigned long long)(size_t)gptr)
                 : "memory");
}
__device__ __forceinline__ void async_wait0() {
    asm volatile("s_wait_asynccnt 0x0" ::: "memory");
}

union FragU { v16bf v; unsigned int u[8]; u32x4 q[2]; };

// A-matrix fragment (16x32 bf16) from row-major [row][k] LDS tile.
// ISA 7.12.2: lane&15 = M; lanes>=16 shift K by 8; V0-3 hold K base 0, V4-7 base 16.
// -> per lane: ushorts [0..7] and [16..23] relative to (row, base) = 2 x b128.
__device__ __forceinline__ v16bf load_a_frag(const unsigned short* p, int stride,
                                             int row, int koff, int lane) {
    FragU f;
    const unsigned short* rp = p + row * stride + koff + ((lane & 16) ? 8 : 0);
    f.q[0] = *(const u32x4*)(rp);
    f.q[1] = *(const u32x4*)(rp + 16);
    return f.v;
}

// B-matrix fragment (32x16 bf16) from "transposed" [n][k] LDS tile.
// lane&15 = N; same K order as the A fragment: ushorts [0..7] and [16..23]
// relative to (row, koff + (lane>=16?8:0)) = 2 x b128.
__device__ __forceinline__ v16bf load_b_frag_t(const unsigned short* p, int stride,
                                               int row, int koff, int lane) {
    FragU f;
    const unsigned short* rp = p + row * stride + koff + ((lane & 16) ? 8 : 0);
    f.q[0] = *(const u32x4*)(rp);
    f.q[1] = *(const u32x4*)(rp + 16);
    return f.v;
}

__device__ __forceinline__ v8f wmma_bf16(v16bf a, v16bf b, v8f c) {
    return __builtin_amdgcn_wmma_f32_16x16x32_bf16(false, a, false, b, (short)0, c,
                                                   false, false);
}

// ---------------------------------------------------------------------------
// fp32 (K x N) -> bf16 transposed (N x K), tiled through LDS.
// ---------------------------------------------------------------------------
__global__ __launch_bounds__(256) void cvt_transpose_bf16(
    const float* __restrict__ in, unsigned short* __restrict__ out, int K, int N)
{
    __shared__ unsigned short t[64][72];     // padded: no bank conflicts
    const int k0 = blockIdx.x * 64, n0 = blockIdx.y * 64;
    const int tid = threadIdx.x;
    const int r = tid >> 4, c = (tid & 15) * 4;
#pragma unroll
    for (int i = 0; i < 4; ++i) {
        int row = r + i * 16;
        f32x4 v = *(const f32x4*)(in + (size_t)(k0 + row) * N + n0 + c);
        t[row][c + 0] = f2bf(v.x); t[row][c + 1] = f2bf(v.y);
        t[row][c + 2] = f2bf(v.z); t[row][c + 3] = f2bf(v.w);
    }
    __syncthreads();
#pragma unroll
    for (int i = 0; i < 4; ++i) {
        int row = r + i * 16;                       // output row (n)
        unsigned short* op = out + (size_t)(n0 + row) * K + k0 + c;
        op[0] = t[c + 0][row]; op[1] = t[c + 1][row];
        op[2] = t[c + 2][row]; op[3] = t[c + 3][row];
    }
}

// ---------------------------------------------------------------------------
// Fused LayerNorm + pack: x -> kvin rows, latents -> qin (+ kvin tail rows).
// ---------------------------------------------------------------------------
__global__ __launch_bounds__(256) void ln_pack(
    const float* __restrict__ x, const float* __restrict__ lat,
    const float* __restrict__ g1, const float* __restrict__ b1,
    const float* __restrict__ g2, const float* __restrict__ b2,
    unsigned short* __restrict__ kvin, unsigned short* __restrict__ qin)
{
    __shared__ float red[256];
    const int row = blockIdx.x, tid = threadIdx.x;
    const float *src, *g, *bb;
    unsigned short *dst0, *dst1 = nullptr;
    if (row < B_X * N_KV) {                       // image features
        int b = row >> 8, t = row & 255;
        src = x + (size_t)row * D_MODEL;
        g = g1; bb = b1;
        dst0 = kvin + ((size_t)b * (N_KV + N_Q) + t) * D_MODEL;
    } else {                                      // latents
        int rl = row - B_X * N_KV;
        int b = rl >> 6, t = rl & 63;
        src = lat + (size_t)rl * D_MODEL;
        g = g2; bb = b2;
        dst0 = qin + (size_t)rl * D_MODEL;
        if (b < B_X)
            dst1 = kvin + ((size_t)b * (N_KV + N_Q) + N_KV + t) * D_MODEL;
    }
    f32x4 v = *(const f32x4*)(src + tid * 4);
    red[tid] = v.x + v.y + v.z + v.w;
    __syncthreads();
    for (int o = 128; o > 0; o >>= 1) { if (tid < o) red[tid] += red[tid + o]; __syncthreads(); }
    float mean = red[0] * (1.0f / D_MODEL);
    __syncthreads();
    float dx = v.x - mean, dy = v.y - mean, dz = v.z - mean, dw = v.w - mean;
    red[tid] = dx * dx + dy * dy + dz * dz + dw * dw;
    __syncthreads();
    for (int o = 128; o > 0; o >>= 1) { if (tid < o) red[tid] += red[tid + o]; __syncthreads(); }
    float inv = rsqrtf(red[0] * (1.0f / D_MODEL) + 1e-5f);
    float d4[4] = {dx, dy, dz, dw};
#pragma unroll
    for (int i = 0; i < 4; ++i) {
        int col = tid * 4 + i;
        unsigned short y = f2bf(d4[i] * inv * g[col] + bb[col]);
        dst0[col] = y;
        if (dst1) dst1[col] = y;
    }
}

// ---------------------------------------------------------------------------
// WMMA GEMM: C[M,N] = alpha * A[M,K] @ B[K,N], with B supplied TRANSPOSED
// (Bt is N x K row-major).  Block tile 64x64, 8 waves, BK = 32.
// Tiles staged via async global->LDS copies (ASYNCcnt).
// mode: 0 = bf16 out, 1 = f32 out, 2 = KV split (n<1024 -> Cv bf16 [M,1024],
//       n>=1024 -> Cv2 = V^T bf16 [1024, M]).
// ---------------------------------------------------------------------------
__global__ __launch_bounds__(256) void gemm_bf16(
    const unsigned short* __restrict__ A, const unsigned short* __restrict__ Bt,
    void* __restrict__ Cv, void* __restrict__ Cv2,
    int M, int N, int K, int lda, int ldb, int ldc, float alpha, int mode)
{
    __shared__ __align__(16) unsigned short As[64 * 32];
    __shared__ __align__(16) unsigned short Bs[64 * 32];   // [n][k]
    const int tid = threadIdx.x;
    const int lane = tid & 31, wave = tid >> 5;
    const int wm = wave & 3, wn = wave >> 2;
    const int bm = blockIdx.y * 64, bn = blockIdx.x * 64;
    const int l15 = lane & 15;
    (void)N;
    v8f acc0 = {}, acc1 = {};
    const int arow = tid >> 2, acol = (tid & 3) << 3;
    const unsigned short* aptr = A  + (size_t)(bm + arow) * lda + acol;
    const unsigned short* bptr = Bt + (size_t)(bn + arow) * ldb + acol;
    const unsigned as_lds = lds_off(&As[arow * 32 + acol]);
    const unsigned bs_lds = lds_off(&Bs[arow * 32 + acol]);
    for (int k0 = 0; k0 < K; k0 += 32) {
        __syncthreads();
        async_copy_b128(as_lds, aptr + k0);
        async_copy_b128(bs_lds, bptr + k0);
        if (k0 + 32 < K) {
            __builtin_prefetch(aptr + k0 + 32, 0, 1);
            __builtin_prefetch(bptr + k0 + 32, 0, 1);
        }
        async_wait0();
        __syncthreads();
        v16bf af  = load_a_frag (As, 32, wm * 16 + l15, 0, lane);
        v16bf bf0 = load_b_frag_t(Bs, 32, wn * 32 + l15, 0, lane);
        v16bf bf1 = load_b_frag_t(Bs, 32, wn * 32 + 16 + l15, 0, lane);
        acc0 = wmma_bf16(af, bf0, acc0);
        acc1 = wmma_bf16(af, bf1, acc1);
    }
    const int mbase = bm + wm * 16 + ((lane & 16) ? 8 : 0);
    const int ncol  = bn + wn * 32 + l15;
#pragma unroll
    for (int r = 0; r < 8; ++r) {
        float v0 = acc0[r] * alpha, v1 = acc1[r] * alpha;
        int gm = mbase + r;
        if (mode == 1) {
            ((float*)Cv)[(size_t)gm * ldc + ncol]      = v0;
            ((float*)Cv)[(size_t)gm * ldc + ncol + 16] = v1;
        } else if (mode == 0) {
            ((unsigned short*)Cv)[(size_t)gm * ldc + ncol]      = f2bf(v0);
            ((unsigned short*)Cv)[(size_t)gm * ldc + ncol + 16] = f2bf(v1);
        } else {                     // KV split: K normal, V transposed
            if (ncol < D_MODEL) {
                ((unsigned short*)Cv)[(size_t)gm * D_MODEL + ncol]      = f2bf(v0);
                ((unsigned short*)Cv)[(size_t)gm * D_MODEL + ncol + 16] = f2bf(v1);
            } else {
                ((unsigned short*)Cv2)[(size_t)(ncol - D_MODEL) * M + gm]      = f2bf(v0);
                ((unsigned short*)Cv2)[(size_t)(ncol + 16 - D_MODEL) * M + gm] = f2bf(v1);
            }
        }
    }
}

// ---------------------------------------------------------------------------
// Flash attention, S^T formulation: one (head, 64-query block) per 128-thread
// workgroup.  S^T = K @ Q^T (K token-major = A operand, Q row-major = B^T),
// so softmax rows live inside a lane; P^T spills to LDS as packed b128;
// O = P @ V with V supplied pre-transposed (V^T tiles stage coalesced).
// K/V chunks staged with async global->LDS copies.
// Mask = KV-range restriction: qb<8 -> rows [qb*320, +320); qb==8 -> all 2560.
// ---------------------------------------------------------------------------
__global__ __launch_bounds__(128) void attn_fwd(
    const unsigned short* __restrict__ Q, const unsigned short* __restrict__ Kp,
    const unsigned short* __restrict__ Vt, unsigned short* __restrict__ O,
    const int* __restrict__ use_mask)
{
    __shared__ __align__(16) unsigned short Qs[64 * 64];   // [q][dh]
    __shared__ __align__(16) unsigned short Ks[64 * 64];   // [token][dh]
    __shared__ __align__(16) unsigned short Vs[64 * 64];   // [dh][token]  (V^T tile)
    __shared__ __align__(16) unsigned short Ps[64 * 64];   // [q][token]
    const int h = blockIdx.x, qb = blockIdx.y;
    const int tid = threadIdx.x, lane = tid & 31, wave = tid >> 5;
    const int l15 = lane & 15;
    const int half8 = (lane & 16) ? 8 : 0;
    const int um = use_mask[0];

    {   // stage Q tile (64 queries x 64 dh)
        int row = tid >> 1, c0 = (tid & 1) * 32;
        const unsigned short* src = Q + (size_t)(qb * 64 + row) * D_MODEL + h * DH + c0;
        unsigned ql = lds_off(&Qs[row * 64 + c0]);
#pragma unroll
        for (int j = 0; j < 4; ++j)
            async_copy_b128(ql + j * 16, src + j * 8);
    }

    v8f o0 = {}, o1 = {}, o2 = {}, o3 = {};
    float m_run = -3.0e38f, l_run = 0.0f;

    const bool diag   = (um != 0) && (qb < B_X);
    const int  nch    = diag ? (N_KV + N_Q) / 64 : ROWS_KV / 64;   // 5 or 40
    const int  kvbase = diag ? qb * (N_KV + N_Q) : 0;

    const int srow = tid >> 1, sc0 = (tid & 1) * 32;
    const unsigned klds = lds_off(&Ks[srow * 64 + sc0]);
    const unsigned vlds = lds_off(&Vs[srow * 64 + sc0]);

    for (int c = 0; c < nch; ++c) {
        const int kv0 = kvbase + c * 64;
        const unsigned short* ksrc = Kp + (size_t)(kv0 + srow) * D_MODEL + h * DH + sc0;
        const unsigned short* vsrc = Vt + (size_t)(h * DH + srow) * ROWS_KV + kv0 + sc0;
        __syncthreads();
#pragma unroll
        for (int j = 0; j < 4; ++j) {
            async_copy_b128(klds + j * 16, ksrc + j * 8);
            async_copy_b128(vlds + j * 16, vsrc + j * 8);
        }
        if (c + 1 < nch) {
            __builtin_prefetch(ksrc + 64 * D_MODEL, 0, 1);   // next token chunk
            __builtin_prefetch(vsrc + 64, 0, 1);
        }
        async_wait0();
        __syncthreads();

        // S^T tiles: rows = tokens (4 tiles of 16), cols = q in [16*wave, +16)
        v8f st[4] = {};
#pragma unroll
        for (int kk = 0; kk < 2; ++kk) {
            v16bf bq = load_b_frag_t(Qs, 64, wave * 16 + l15, kk * 32, lane);
#pragma unroll
            for (int jm = 0; jm < 4; ++jm) {
                v16bf ak = load_a_frag(Ks, 64, jm * 16 + l15, kk * 32, lane);
                st[jm] = wmma_bf16(ak, bq, st[jm]);
            }
        }

        // per-lane online softmax for q = wave*16 + l15 (tokens in registers)
        float mx = -3.0e38f;
#pragma unroll
        for (int jm = 0; jm < 4; ++jm)
#pragma unroll
            for (int r = 0; r < 8; ++r)
                mx = fmaxf(mx, st[jm][r]);
        mx = fmaxf(mx, __shfl_xor(mx, 16, 32));
        float mnew = fmaxf(m_run, mx);
        float corr = __expf(m_run - mnew);
        m_run = mnew;
        float psum = 0.0f;
#pragma unroll
        for (int jm = 0; jm < 4; ++jm) {        // P^T -> LDS, packed b128 per tile
            unsigned int pk[4];
#pragma unroll
            for (int r = 0; r < 4; ++r) {
                float p0 = __expf(st[jm][2 * r]     - mnew);
                float p1 = __expf(st[jm][2 * r + 1] - mnew);
                psum += p0 + p1;
                pk[r] = (unsigned)f2bf(p0) | ((unsigned)f2bf(p1) << 16);
            }
            u32x4 pv; pv.x = pk[0]; pv.y = pk[1]; pv.z = pk[2]; pv.w = pk[3];
            *(u32x4*)(&Ps[(wave * 16 + l15) * 64 + jm * 16 + half8]) = pv;
        }
        psum += __shfl_xor(psum, 16, 32);
        l_run = l_run * corr + psum;

        // rescale O rows (row q' = r + half8 needs corr held by lane r + half8)
#pragma unroll
        for (int r = 0; r < 8; ++r) {
            float cr = __shfl(corr, r + half8, 32);
            o0[r] *= cr; o1[r] *= cr; o2[r] *= cr; o3[r] *= cr;
        }

        // O += P @ V   (Ps is wave-private: same-wave LDS ops are in order)
#pragma unroll
        for (int kk = 0; kk < 2; ++kk) {
            v16bf ap = load_a_frag(Ps, 64, wave * 16 + l15, kk * 32, lane);
            v16bf b0 = load_b_frag_t(Vs, 64, 0 * 16 + l15, kk * 32, lane);
            v16bf b1 = load_b_frag_t(Vs, 64, 1 * 16 + l15, kk * 32, lane);
            v16bf b2 = load_b_frag_t(Vs, 64, 2 * 16 + l15, kk * 32, lane);
            v16bf b3 = load_b_frag_t(Vs, 64, 3 * 16 + l15, kk * 32, lane);
            o0 = wmma_bf16(ap, b0, o0);
            o1 = wmma_bf16(ap, b1, o1);
            o2 = wmma_bf16(ap, b2, o2);
            o3 = wmma_bf16(ap, b3, o3);
        }
    }

    // epilogue: O / l -> attn_out bf16, layout (b*n_q, H*DH)
    float linv = 1.0f / l_run;
#pragma unroll
    for (int r = 0; r < 8; ++r) {
        float lr = __shfl(linv, r + half8, 32);
        int gq = qb * 64 + wave * 16 + r + half8;
        size_t base = (size_t)gq * D_MODEL + h * DH;
        O[base + 0 * 16 + l15] = f2bf(o0[r] * lr);
        O[base + 1 * 16 + l15] = f2bf(o1[r] * lr);
        O[base + 2 * 16 + l15] = f2bf(o2[r] * lr);
        O[base + 3 * 16 + l15] = f2bf(o3[r] * lr);
    }
}

// ---------------------------------------------------------------------------
extern "C" void kernel_launch(void* const* d_in, const int* in_sizes, int n_in,
                              void* d_out, int out_size, void* d_ws, size_t ws_size,
                              hipStream_t stream)
{
    (void)in_sizes; (void)n_in; (void)out_size; (void)ws_size;
    const float* x    = (const float*)d_in[0];
    const float* lat  = (const float*)d_in[1];
    const float* g1   = (const float*)d_in[2];
    const float* b1   = (const float*)d_in[3];
    const float* g2   = (const float*)d_in[4];
    const float* b2   = (const float*)d_in[5];
    const float* Wq   = (const float*)d_in[6];
    const float* Wkv  = (const float*)d_in[7];
    const float* Wout = (const float*)d_in[8];
    const int* use_mask = (const int*)d_in[9];

    char* ws = (char*)d_ws;
    size_t off = 0;
    auto take = [&](size_t bytes) {
        char* p = ws + off;
        off += (bytes + 255) & ~(size_t)255;
        return p;
    };
    unsigned short* qin   = (unsigned short*)take((size_t)ROWS_Q  * D_MODEL * 2);
    unsigned short* kvin  = (unsigned short*)take((size_t)ROWS_KV * D_MODEL * 2);
    unsigned short* wqt   = (unsigned short*)take((size_t)D_MODEL * D_MODEL * 2);
    unsigned short* wkvt  = (unsigned short*)take((size_t)2 * D_MODEL * D_MODEL * 2);
    unsigned short* woutt = (unsigned short*)take((size_t)D_MODEL * D_MODEL * 2);
    unsigned short* qp    = (unsigned short*)take((size_t)ROWS_Q  * D_MODEL * 2);
    unsigned short* kp    = (unsigned short*)take((size_t)ROWS_KV * D_MODEL * 2);
    unsigned short* vtp   = (unsigned short*)take((size_t)D_MODEL * ROWS_KV * 2);
    unsigned short* ao    = (unsigned short*)take((size_t)ROWS_Q  * D_MODEL * 2);

    // 1) weights -> transposed bf16 (N x K)
    cvt_transpose_bf16<<<dim3(16, 16), 256, 0, stream>>>(Wq,   wqt,   D_MODEL, D_MODEL);
    cvt_transpose_bf16<<<dim3(16, 32), 256, 0, stream>>>(Wkv,  wkvt,  D_MODEL, 2 * D_MODEL);
    cvt_transpose_bf16<<<dim3(16, 16), 256, 0, stream>>>(Wout, woutt, D_MODEL, D_MODEL);

    // 2) layernorm + pack
    ln_pack<<<dim3(B_X * N_KV + ROWS_Q), 256, 0, stream>>>(x, lat, g1, b1, g2, b2, kvin, qin);

    // 3) Q = 0.125 * qin @ Wq     (DH^-0.5 folded into Q)
    gemm_bf16<<<dim3(D_MODEL / 64, ROWS_Q / 64), 256, 0, stream>>>(
        qin, wqt, qp, nullptr, ROWS_Q, D_MODEL, D_MODEL,
        D_MODEL, D_MODEL, D_MODEL, 0.125f, 0);

    // 4) KV = kvin @ Wkv  ->  K [2560,1024] bf16  +  V^T [1024,2560] bf16
    gemm_bf16<<<dim3(2 * D_MODEL / 64, ROWS_KV / 64), 256, 0, stream>>>(
        kvin, wkvt, kp, vtp, ROWS_KV, 2 * D_MODEL, D_MODEL,
        D_MODEL, D_MODEL, D_MODEL, 1.0f, 2);

    // 5) attention
    attn_fwd<<<dim3(HEADS, B_LAT), 128, 0, stream>>>(qp, kp, vtp, ao, use_mask);

    // 6) out = ao @ Wout -> d_out (fp32)
    gemm_bf16<<<dim3(D_MODEL / 64, ROWS_Q / 64), 256, 0, stream>>>(
        ao, woutt, d_out, nullptr, ROWS_Q, D_MODEL, D_MODEL,
        D_MODEL, D_MODEL, D_MODEL, 1.0f, 1);
}
